// LSTMPredictor_44435731644904
// MI455X (gfx1250) — matrix operand order repaired and checked by reading, not compiled, not run on hardware
//
#include <hip/hip_runtime.h>
#include <hip/hip_bf16.h>
#include <math.h>

// Problem constants (from reference)
#define D_    80
#define H_    178
#define FH_   712      // 4*H
#define T_    262144
#define NT_   45       // ceil(712/16) N-tiles
#define KT_   3        // ceil(80/32)  K-tiles
#define SWS_  182      // LDS row stride (halfs); dword-stride 91 is odd -> good bank spread

typedef __attribute__((ext_vector_type(16))) _Float16 v16h;
typedef __attribute__((ext_vector_type(8)))  float    v8f;

// ---------------------------------------------------------------------------
// Kernel 0: pack W_ih [712,80] fp32 into WMMA-ready B fragments (f16).
// B-matrix 32x16 (16-bit) lane layout used here (contiguous k, not the ISA table order):
//   VGPR v, lane: N = lane%16, K = 16*(lane/16) + 2*v + j   (j = half within dword)
// Fragment (nt, kt): B[k, n] = W_ih[nt*16+n, kt*32+k], zero-padded.
// Stored as 16 contiguous halfs per lane -> single 32B load in the GEMM.
// ---------------------------------------------------------------------------
__global__ void pack_wih_kernel(const float* __restrict__ W_ih,
                                _Float16* __restrict__ Bpack) {
  const int lane = threadIdx.x;           // 0..31
  const int nt   = blockIdx.x;            // 0..44
  const int kt   = blockIdx.y;            // 0..2
  const int lh   = lane >> 4;
  const int col  = nt * 16 + (lane & 15); // N (row of W_ih)
  _Float16 vals[16];
#pragma unroll
  for (int v = 0; v < 8; ++v) {
#pragma unroll
    for (int j = 0; j < 2; ++j) {
      const int K = 16 * lh + 2 * v + j;
      const int k = kt * 32 + K;
      float x = 0.0f;
      if (col < FH_ && k < D_) x = W_ih[(size_t)col * D_ + k];
      vals[2 * v + j] = (_Float16)x;
    }
  }
  const size_t base = ((size_t)(nt * KT_ + kt) * 32 + lane) * 16;
#pragma unroll
  for (int s = 0; s < 16; ++s) Bpack[base + s] = vals[s];
}

// ---------------------------------------------------------------------------
// Kernel 1: x_gates[T,712] = feats[T,80] @ W_ih^T + (b_ih + b_hh), via WMMA.
// One block = one 16-row M tile; 8 waves split the 45 N tiles.
// A-matrix 16x32 (16-bit) lane layout used here (contiguous k, same as B pack):
//   lane: M = lane%16; VGPR v: K = 16*(lane/16)+2v+j
// C/D 16x16 f32: VGPR r -> M = r + 8*(lane/16), N = lane%16.
// ---------------------------------------------------------------------------
__global__ __launch_bounds__(256) void xgates_gemm_kernel(
    const float* __restrict__ feats, const _Float16* __restrict__ Bpack,
    const float* __restrict__ b_ih, const float* __restrict__ b_hh,
    float* __restrict__ xg) {
  __shared__ float sfeat[16 * D_];                 // [m][k] fp32 tile
  const int tid = threadIdx.x;
  const int mb  = blockIdx.x * 16;
  for (int i = tid; i < 16 * D_; i += 256) {
    const int m = i / D_, k = i % D_;
    sfeat[i] = feats[(size_t)(mb + m) * D_ + k];
  }
  __syncthreads();

  const int wave = tid >> 5, lane = tid & 31;
  const int lh = lane >> 4, ll = lane & 15;

  // Build the three A K-fragments once per wave (rows M = ll of this tile).
  v16h a[KT_];
#pragma unroll
  for (int kt = 0; kt < KT_; ++kt) {
#pragma unroll
    for (int v = 0; v < 8; ++v) {
#pragma unroll
      for (int j = 0; j < 2; ++j) {
        const int K = 16 * lh + 2 * v + j;
        const int k = kt * 32 + K;
        const float x = (k < D_) ? sfeat[ll * D_ + k] : 0.0f;
        a[kt][2 * v + j] = (_Float16)x;
      }
    }
  }

  for (int nt = wave; nt < NT_; nt += 8) {
    v8f c = {};
#pragma unroll
    for (int kt = 0; kt < KT_; ++kt) {
      const v16h b =
          *(const v16h*)(Bpack + ((size_t)(nt * KT_ + kt) * 32 + lane) * 16);
      c = __builtin_amdgcn_wmma_f32_16x16x32_f16(
          /*neg_a=*/false, a[kt], /*neg_b=*/false, b,
          /*c_mod=*/(short)0, c, /*reuse_a=*/false, /*reuse_b=*/false);
    }
    const int col = nt * 16 + ll;
    if (col < FH_) {
      const float bias = b_ih[col] + b_hh[col];
#pragma unroll
      for (int r = 0; r < 8; ++r) {
        const int row = mb + r + 8 * lh;
        xg[(size_t)row * FH_ + col] = c[r] + bias;
      }
    }
  }
}

// ---------------------------------------------------------------------------
// Kernel 2: the sequential scan. ONE persistent workgroup, 1024 threads
// (32 wave32 waves). W_hh + C live in LDS as f16 (293.8 KB of the 320 KB WGP
// LDS). Per step:
//   wave0: scale = max(||x_t||, eps)        (shfl reduction)
//   threads 0..711 : gates row dot  (xg_t + W_hh @ h_prev)
//   threads 712..791: C @ h_prev -> pred, err partial
//   threads 0..177 : LSTM elementwise update; wave6: err shfl reduction
// x_gates / feats rows are register double-buffered (load t+1 while computing
// t) and prefetched 8 steps ahead via global_prefetch_b8.
// ---------------------------------------------------------------------------
__global__ __launch_bounds__(1024) void lstm_scan_kernel(
    const float* __restrict__ feats, const float* __restrict__ Cmat,
    const float* __restrict__ W_hh, const float* __restrict__ xg,
    float* __restrict__ preds, float* __restrict__ errs) {
  extern __shared__ char smem[];
  _Float16* sW  = (_Float16*)smem;                       // [712][182]
  _Float16* sCl = sW + (size_t)FH_ * SWS_;               // [80][182]
  float* sh     = (float*)(sCl + (size_t)D_ * SWS_);     // h[178] (+pad)
  float* sc     = sh + 180;                              // c[178] (+pad)
  float* sg     = sc + 180;                              // raw gates [712]
  float* sx     = sg + FH_;                              // x_t [80]
  float* sxp    = sx + D_;                               // x_pred [80]
  float* sred   = sxp + D_;                              // err partials [128]
  float* sscale = sred + 128;                            // [0]=scale

  const int tid = threadIdx.x;

  // Stage recurrent weights into LDS as f16.
  for (int i = tid; i < FH_ * H_; i += 1024) {
    const int r = i / H_, k = i - r * H_;
    sW[(size_t)r * SWS_ + k] = (_Float16)W_hh[i];
  }
  for (int i = tid; i < D_ * H_; i += 1024) {
    const int r = i / H_, k = i - r * H_;
    sCl[(size_t)r * SWS_ + k] = (_Float16)Cmat[i];
  }
  if (tid < H_) { sh[tid] = 0.0f; sc[tid] = 0.0f; }
  if (tid < 128) sred[tid] = 0.0f;
  __syncthreads();

  // Register double buffers for the per-step global streams.
  float xg_cur = (tid < FH_) ? xg[tid] : 0.0f;
  float x_cur  = (tid < D_) ? feats[tid] : 0.0f;

  for (int t = 0; t < T_; ++t) {
    // Issue next-step loads early; they retire behind this step's compute.
    float xg_nxt = 0.0f, x_nxt = 0.0f;
    if (t + 1 < T_) {
      if (tid < FH_) xg_nxt = xg[(size_t)(t + 1) * FH_ + tid];
      if (tid < D_)  x_nxt  = feats[(size_t)(t + 1) * D_ + tid];
    }
    if (t + 8 < T_) {
      if (tid < FH_) __builtin_prefetch(&xg[(size_t)(t + 8) * FH_ + tid], 0, 1);
      if (tid < D_)  __builtin_prefetch(&feats[(size_t)(t + 8) * D_ + tid], 0, 1);
    }

    if (tid < D_) sx[tid] = x_cur;
    __syncthreads();

    // scale = max(||x_t||, eps)  — wave 0 reduction
    if (tid < 32) {
      float s = sx[tid] * sx[tid] + sx[tid + 32] * sx[tid + 32];
      if (tid < 16) s += sx[tid + 64] * sx[tid + 64];
      for (int off = 16; off; off >>= 1) s += __shfl_down(s, off, 32);
      if (tid == 0) sscale[0] = fmaxf(sqrtf(s), 1e-6f);
    }
    __syncthreads();
    const float scale = sscale[0];

    // Row dot products against h_prev (still in sh).
    if (tid < FH_) {
      float acc = xg_cur;
      const _Float16* w = sW + (size_t)tid * SWS_;
#pragma unroll 2
      for (int k = 0; k < 176; k += 4) {
        acc = fmaf((float)w[k],     sh[k],     acc);
        acc = fmaf((float)w[k + 1], sh[k + 1], acc);
        acc = fmaf((float)w[k + 2], sh[k + 2], acc);
        acc = fmaf((float)w[k + 3], sh[k + 3], acc);
      }
      acc = fmaf((float)w[176], sh[176], acc);
      acc = fmaf((float)w[177], sh[177], acc);
      sg[tid] = acc;
    } else if (tid < FH_ + D_) {
      const int r = tid - FH_;
      float acc = 0.0f;
      const _Float16* w = sCl + (size_t)r * SWS_;
#pragma unroll 2
      for (int k = 0; k < 176; k += 4) {
        acc = fmaf((float)w[k],     sh[k],     acc);
        acc = fmaf((float)w[k + 1], sh[k + 1], acc);
        acc = fmaf((float)w[k + 2], sh[k + 2], acc);
        acc = fmaf((float)w[k + 3], sh[k + 3], acc);
      }
      acc = fmaf((float)w[176], sh[176], acc);
      acc = fmaf((float)w[177], sh[177], acc);
      const float xp = tanhf(acc) * scale;
      sxp[r] = xp;
      preds[(size_t)t * D_ + r] = xp;
      const float e = sx[r] - xp;
      sred[r] = e * e;
    }
    __syncthreads();

    // LSTM elementwise update (threads 0..177) + err reduction (wave 6).
    if (tid < H_) {
      const float gi = sg[tid];
      const float gf = sg[H_ + tid];
      const float gg = sg[2 * H_ + tid];
      const float go = sg[3 * H_ + tid];
      const float i_s = 1.0f / (1.0f + __expf(-gi));
      const float f_s = 1.0f / (1.0f + __expf(-gf));
      const float g_t = tanhf(gg);
      const float o_s = 1.0f / (1.0f + __expf(-go));
      const float cn = f_s * sc[tid] + i_s * g_t;
      sc[tid] = cn;
      sh[tid] = o_s * tanhf(cn);
    } else if (tid >= 192 && tid < 224) {
      const int l = tid - 192;
      float s = sred[l] + sred[l + 32];
      if (l < 16) s += sred[l + 64];
      for (int off = 16; off; off >>= 1) s += __shfl_down(s, off, 32);
      if (l == 0) errs[t] = fminf(sqrtf(s) / scale, 4.0f);
    }
    __syncthreads();

    xg_cur = xg_nxt;
    x_cur  = x_nxt;
  }
}

// ---------------------------------------------------------------------------
// Launch. Workspace layout:
//   [0, XG_BYTES)          x_gates fp32 [T,712]   (746.6 MB)
//   [XG_BYTES, +138240)    packed W_ih f16 fragments
// ---------------------------------------------------------------------------
extern "C" void kernel_launch(void* const* d_in, const int* in_sizes, int n_in,
                              void* d_out, int out_size, void* d_ws,
                              size_t ws_size, hipStream_t stream) {
  (void)in_sizes; (void)n_in; (void)out_size; (void)ws_size;
  const float* feats = (const float*)d_in[0];
  const float* Cmat  = (const float*)d_in[1];
  const float* W_ih  = (const float*)d_in[2];
  const float* W_hh  = (const float*)d_in[3];
  const float* b_ih  = (const float*)d_in[4];
  const float* b_hh  = (const float*)d_in[5];

  float* preds = (float*)d_out;                    // [T, 80]
  float* errsv = preds + (size_t)T_ * D_;          // [T]

  const size_t XG_BYTES = (size_t)T_ * FH_ * sizeof(float);
  float*    xg    = (float*)d_ws;
  _Float16* Bpack = (_Float16*)((char*)d_ws + XG_BYTES);

  pack_wih_kernel<<<dim3(NT_, KT_), 32, 0, stream>>>(W_ih, Bpack);
  xgates_gemm_kernel<<<T_ / 16, 256, 0, stream>>>(feats, Bpack, b_ih, b_hh, xg);

  const size_t smem_bytes =
      (size_t)(FH_ + D_) * SWS_ * sizeof(_Float16) +
      (180 + 180 + FH_ + D_ + D_ + 128 + 8) * sizeof(float);
  lstm_scan_kernel<<<1, 1024, smem_bytes, stream>>>(feats, Cmat, W_hh, xg,
                                                    preds, errsv);
}
